// VAE_41936060678441
// MI455X (gfx1250) — matrix operand order repaired and checked by reading, not compiled, not run on hardware
//
#include <hip/hip_runtime.h>
#include <cstddef>

// ---------------- model dims ----------------
#define BB   256    // batch
#define TT   96     // timesteps
#define NIN  256    // input feature dim
#define HH   1024   // GRU hidden
#define G3   3072   // 3*HH gate dim
#define LATD 512    // latent
#define EHD  4096   // 2*LAYERS*HH encoder concat
#define NG   5

// ---------------- WMMA types ----------------
typedef __attribute__((ext_vector_type(4)))  unsigned int u32x4;
typedef __attribute__((ext_vector_type(16))) __bf16       v16bf;
typedef __attribute__((ext_vector_type(8)))  float        v8f;

union Frag32 { u32x4 q[2]; v16bf h; };

__device__ __forceinline__ unsigned short f2bf(float f) {
  unsigned int u = __float_as_uint(f);
  u += 0x7FFFu + ((u >> 16) & 1u);   // round-to-nearest-even
  return (unsigned short)(u >> 16);
}

// -----------------------------------------------------------------------------
// Core: C[64,N-tile 64] block = act( A[M,K](bf16) @ W[N,K]^T(bf16) + bias ).
// One wave computes a 16(M) x 64(N) tile; 4 waves/block stack along M.
// Software-pipelined: two register fragment buffers (ping/pong) so the
// s_wait before each WMMA leaves the next k-step's loads in flight.
// Requires K % 64 == 0 and K >= 64 (true for all GEMMs here).
// -----------------------------------------------------------------------------
__device__ __forceinline__ void gemm_core(
    const unsigned short* __restrict__ A, int lda,
    const unsigned short* __restrict__ W, int K,
    const float* __restrict__ bias,
    float* __restrict__ C, int ldc,
    unsigned short* __restrict__ Cb, int ldcb, int act)
{
  const int lane  = threadIdx.x & 31;
  const int wave  = threadIdx.x >> 5;
  const int kg    = lane >> 4;
  const int l16   = lane & 15;
  const int mTile = blockIdx.y * 64 + wave * 16;
  const int nTile = blockIdx.x * 64;

  const unsigned short* arow = A + (size_t)(mTile + l16) * lda;
  const unsigned short* wrow[4];
#pragma unroll
  for (int t = 0; t < 4; ++t)
    wrow[t] = W + (size_t)(nTile + 16 * t + l16) * (size_t)K + 16 * kg;

  auto ldA = [&](Frag32& f, int k0) {
    f.q[0] = *(const u32x4*)(arow + k0 + 16 * kg);
    f.q[1] = *(const u32x4*)(arow + k0 + 16 * kg + 8);
  };
  auto ldB = [&](Frag32* f, int k0) {
#pragma unroll
    for (int t = 0; t < 4; ++t) {
      f[t].q[0] = *(const u32x4*)(wrow[t] + k0);
      f[t].q[1] = *(const u32x4*)(wrow[t] + k0 + 8);
    }
  };

  const v8f vzero = {0.f,0.f,0.f,0.f,0.f,0.f,0.f,0.f};
  v8f acc[4];
  acc[0] = vzero; acc[1] = vzero; acc[2] = vzero; acc[3] = vzero;

  Frag32 aA, aB, bA[4], bB[4];
  ldA(aA, 0);  ldB(bA, 0);
  ldA(aB, 32); ldB(bB, 32);

  int k0 = 0;
  for (; k0 + 64 < K; k0 += 64) {
#pragma unroll
    for (int t = 0; t < 4; ++t)
      acc[t] = __builtin_amdgcn_wmma_f32_16x16x32_bf16(
          false, aA.h, false, bA[t].h, (short)0, acc[t], false, false);
    ldA(aA, k0 + 64); ldB(bA, k0 + 64);      // prefetch while aB/bB consumed next
#pragma unroll
    for (int t = 0; t < 4; ++t)
      acc[t] = __builtin_amdgcn_wmma_f32_16x16x32_bf16(
          false, aB.h, false, bB[t].h, (short)0, acc[t], false, false);
    ldA(aB, k0 + 96); ldB(bB, k0 + 96);
  }
  // tail: both buffers hold the final two k-steps
#pragma unroll
  for (int t = 0; t < 4; ++t)
    acc[t] = __builtin_amdgcn_wmma_f32_16x16x32_bf16(
        false, aA.h, false, bA[t].h, (short)0, acc[t], false, false);
#pragma unroll
  for (int t = 0; t < 4; ++t)
    acc[t] = __builtin_amdgcn_wmma_f32_16x16x32_bf16(
        false, aB.h, false, bB[t].h, (short)0, acc[t], false, false);

#pragma unroll
  for (int t = 0; t < 4; ++t) {
    const int col = nTile + 16 * t + l16;
    const float bv = bias ? bias[col] : 0.0f;
#pragma unroll
    for (int r = 0; r < 8; ++r) {
      const int row = mTile + r + 8 * kg;   // C/D: vgpr r -> M=r (lanes 0-15), M=r+8 (16-31)
      float v = acc[t][r] + bv;
      if (act) v = (v > 0.0f) ? v : 0.01f * v;  // leaky relu
      C[(size_t)row * ldc + col] = v;
      if (Cb) Cb[(size_t)row * ldcb + col] = f2bf(v);
    }
  }
}

__global__ __launch_bounds__(128) void wmma_gemm_k(
    const unsigned short* __restrict__ A, int lda,
    const unsigned short* __restrict__ W, int K,
    const float* __restrict__ bias,
    float* __restrict__ C, int ldc,
    unsigned short* __restrict__ Cb, int ldcb, int act)
{
  gemm_core(A, lda, W, K, bias, C, ldc, Cb, ldcb, act);
}

// Two independent GEMMs (gi and gh gate matmuls) fused into one launch;
// blockIdx.z selects the parameter set (scalar-uniform -> EXEC stays full).
__global__ __launch_bounds__(128) void wmma_gemm2_k(
    const unsigned short* __restrict__ A0, int lda0,
    const unsigned short* __restrict__ W0, int K0,
    const float* __restrict__ b0, float* __restrict__ C0,
    const unsigned short* __restrict__ A1, int lda1,
    const unsigned short* __restrict__ W1, int K1,
    const float* __restrict__ b1, float* __restrict__ C1,
    int ldc)
{
  if (blockIdx.z == 0) gemm_core(A0, lda0, W0, K0, b0, C0, ldc, nullptr, 0, 0);
  else                 gemm_core(A1, lda1, W1, K1, b1, C1, ldc, nullptr, 0, 0);
}

// ---------------- pointwise kernels ----------------
__global__ __launch_bounds__(256) void f32_to_bf16_k(
    const float* __restrict__ s, unsigned short* __restrict__ d, int n) {
  int i = blockIdx.x * blockDim.x + threadIdx.x;
  if (i < n) d[i] = f2bf(s[i]);
}

__global__ __launch_bounds__(256) void zero_h_k(
    float* __restrict__ h, unsigned short* __restrict__ hb, int n) {
  int i = blockIdx.x * blockDim.x + threadIdx.x;
  if (i < n) { h[i] = 0.0f; hb[i] = 0; }
}

// h' = (1-z)*tanh(inn + r*hn) + z*h ; gates from gi(+b_ih folded), gh(+b_hh folded)
__global__ __launch_bounds__(256) void gru_pw_k(
    const float* __restrict__ gi, const float* __restrict__ gh,
    float* __restrict__ h, unsigned short* __restrict__ hb) {
  int i = blockIdx.x * blockDim.x + threadIdx.x;
  if (i >= BB * HH) return;
  int b = i >> 10, j = i & (HH - 1);
  const float* gib = gi + (size_t)b * G3;
  const float* ghb = gh + (size_t)b * G3;
  float r  = 1.0f / (1.0f + __expf(-(gib[j]          + ghb[j])));
  float zg = 1.0f / (1.0f + __expf(-(gib[HH + j]     + ghb[HH + j])));
  float n  = tanhf(gib[2 * HH + j] + r * ghb[2 * HH + j]);
  float out = (1.0f - zg) * n + zg * h[i];
  h[i]  = out;
  hb[i] = f2bf(out);
}

__global__ __launch_bounds__(256) void concat_hidden_k(
    const unsigned short* __restrict__ h0, const unsigned short* __restrict__ h1,
    const unsigned short* __restrict__ h2, const unsigned short* __restrict__ h3,
    unsigned short* __restrict__ outb) {
  int i = blockIdx.x * blockDim.x + threadIdx.x;
  if (i >= BB * EHD) return;
  int b = i >> 12, c = i & (EHD - 1);
  int sel = c >> 10, j = c & (HH - 1);
  const unsigned short* src = (sel == 0) ? h0 : (sel == 1) ? h1 : (sel == 2) ? h2 : h3;
  outb[i] = src[(b << 10) + j];
}

__global__ __launch_bounds__(256) void note0_k(
    const float* __restrict__ x, float* __restrict__ out,
    unsigned short* __restrict__ noteb) {
  int i = blockIdx.x * blockDim.x + threadIdx.x;
  if (i >= BB * NIN) return;
  int b = i >> 8, c = i & (NIN - 1);
  float v = x[(size_t)b * TT * NIN + c];
  out[(size_t)b * TT * NIN + c] = v;
  noteb[i] = f2bf(v);
}

__global__ __launch_bounds__(256) void z_k(
    const float* __restrict__ mu, const float* __restrict__ lv,
    const float* __restrict__ eps, float* __restrict__ zf,
    unsigned short* __restrict__ zb) {
  int i = blockIdx.x * blockDim.x + threadIdx.x;
  if (i >= BB * LATD) return;
  float z = mu[i] + __expf(0.5f * lv[i]) * eps[i];
  zf[i] = z; zb[i] = f2bf(z);
}

__global__ __launch_bounds__(256) void genre_k(
    const float* __restrict__ zf, float* __restrict__ out) {
  int b = blockIdx.x * blockDim.x + threadIdx.x;
  if (b >= BB) return;
  float v[NG]; float m = -3.4e38f;
  for (int g = 0; g < NG; ++g) { v[g] = zf[b * LATD + g]; m = fmaxf(m, v[g]); }
  float s = 0.f;
  for (int g = 0; g < NG; ++g) { v[g] = __expf(v[g] - m); s += v[g]; }
  float inv = 1.0f / s;
  for (int g = 0; g < NG; ++g) out[b * NG + g] = v[g] * inv;
}

// ---------------- host orchestration ----------------
extern "C" void kernel_launch(void* const* d_in, const int* in_sizes, int n_in,
                              void* d_out, int out_size, void* d_ws, size_t ws_size,
                              hipStream_t stream) {
  (void)in_sizes; (void)out_size;
  if (n_in < 46) return;

  const float* x   = (const float*)d_in[0];
  const float* eps = (const float*)d_in[1];

  // jax pytree order: params keys sorted (benc, dec, enc, fc_dec, fc_latent,
  // fc_logvar, fc_mu); gru keys sorted (b_hh, b_ih, w_hh, w_ih); lin (W, b).
  struct GruP { const float *b_hh, *b_ih, *w_hh, *w_ih; };
  auto G = [&](int base) -> GruP {
    return GruP{ (const float*)d_in[base],     (const float*)d_in[base + 1],
                 (const float*)d_in[base + 2], (const float*)d_in[base + 3] };
  };
  GruP benc0 = G(2),  benc1 = G(6);
  GruP dec0  = G(10), dec1  = G(14);
  GruP enc0  = G(18), enc1  = G(22);
  const float* fcdW0 = (const float*)d_in[26]; const float* fcdB0 = (const float*)d_in[27];
  const float* fcdW1 = (const float*)d_in[28]; const float* fcdB1 = (const float*)d_in[29];
  const float* latW[2][2]; const float* latB[2][2];
  latW[0][0] = (const float*)d_in[30]; latB[0][0] = (const float*)d_in[31];
  latW[0][1] = (const float*)d_in[32]; latB[0][1] = (const float*)d_in[33];
  latW[1][0] = (const float*)d_in[34]; latB[1][0] = (const float*)d_in[35];
  latW[1][1] = (const float*)d_in[36]; latB[1][1] = (const float*)d_in[37];
  const float* lvW0 = (const float*)d_in[38]; const float* lvB0 = (const float*)d_in[39];
  const float* lvW1 = (const float*)d_in[40]; const float* lvB1 = (const float*)d_in[41];
  const float* muW0 = (const float*)d_in[42]; const float* muB0 = (const float*)d_in[43];
  const float* muW1 = (const float*)d_in[44]; const float* muB1 = (const float*)d_in[45];

  // ---- workspace bump allocator ----
  char* ws = (char*)d_ws;
  size_t off = 0;
  auto aus = [&](size_t n) -> unsigned short* {
    off = (off + 255) & ~(size_t)255;
    unsigned short* p = (unsigned short*)(ws + off);
    off += n * sizeof(unsigned short); return p;
  };
  auto af = [&](size_t n) -> float* {
    off = (off + 255) & ~(size_t)255;
    float* p = (float*)(ws + off);
    off += n * sizeof(float); return p;
  };

  // bf16 weight mirrors (converted once per launch; resident in L2 for reuse)
  unsigned short *benc0_wih = aus((size_t)G3 * NIN), *benc0_whh = aus((size_t)G3 * HH);
  unsigned short *benc1_wih = aus((size_t)G3 * HH),  *benc1_whh = aus((size_t)G3 * HH);
  unsigned short *dec0_wih  = aus((size_t)G3 * NIN), *dec0_whh  = aus((size_t)G3 * HH);
  unsigned short *dec1_wih  = aus((size_t)G3 * HH),  *dec1_whh  = aus((size_t)G3 * HH);
  unsigned short *enc0_wih  = aus((size_t)G3 * NIN), *enc0_whh  = aus((size_t)G3 * HH);
  unsigned short *enc1_wih  = aus((size_t)G3 * HH),  *enc1_whh  = aus((size_t)G3 * HH);
  unsigned short *muW0b = aus((size_t)EHD * EHD), *muW1b = aus((size_t)LATD * EHD);
  unsigned short *lvW0b = aus((size_t)EHD * EHD), *lvW1b = aus((size_t)LATD * EHD);
  unsigned short *latW0b[2], *latW1b[2];
  latW0b[0] = aus((size_t)LATD * LATD); latW1b[0] = aus((size_t)HH * LATD);
  latW0b[1] = aus((size_t)LATD * LATD); latW1b[1] = aus((size_t)HH * LATD);
  unsigned short *fcdW0b = aus((size_t)HH * HH), *fcdW1b = aus((size_t)NIN * HH);
  unsigned short *xb     = aus((size_t)BB * TT * NIN);

  // activations
  float *gi = af((size_t)BB * G3), *gh = af((size_t)BB * G3);
  float* hf[6]; unsigned short* hb[6];
  for (int i = 0; i < 6; ++i) { hf[i] = af((size_t)BB * HH); hb[i] = aus((size_t)BB * HH); }
  unsigned short* hiddenb = aus((size_t)BB * EHD);
  float* fcmid = af((size_t)BB * EHD); unsigned short* fcmidb = aus((size_t)BB * EHD);
  float* zf = af((size_t)BB * LATD);   unsigned short* zb = aus((size_t)BB * LATD);
  float* latmid = af((size_t)BB * LATD); unsigned short* latmidb = aus((size_t)BB * LATD);
  unsigned short* noteb = aus((size_t)BB * NIN);
  float* fcdmid = af((size_t)BB * HH); unsigned short* fcdmidb = aus((size_t)BB * HH);

  if (off > ws_size) return;  // not enough scratch; bail without touching memory

  float* out    = (float*)d_out;
  float* out_mu = out + (size_t)BB * TT * NIN;
  float* out_lv = out_mu + (size_t)BB * LATD;
  float* out_g  = out_lv + (size_t)BB * LATD;

  auto conv = [&](const float* s, unsigned short* d, size_t n) {
    f32_to_bf16_k<<<dim3((unsigned)((n + 255) / 256)), dim3(256), 0, stream>>>(s, d, (int)n);
  };
  conv(benc0.w_ih, benc0_wih, (size_t)G3 * NIN); conv(benc0.w_hh, benc0_whh, (size_t)G3 * HH);
  conv(benc1.w_ih, benc1_wih, (size_t)G3 * HH);  conv(benc1.w_hh, benc1_whh, (size_t)G3 * HH);
  conv(dec0.w_ih,  dec0_wih,  (size_t)G3 * NIN); conv(dec0.w_hh,  dec0_whh,  (size_t)G3 * HH);
  conv(dec1.w_ih,  dec1_wih,  (size_t)G3 * HH);  conv(dec1.w_hh,  dec1_whh,  (size_t)G3 * HH);
  conv(enc0.w_ih,  enc0_wih,  (size_t)G3 * NIN); conv(enc0.w_hh,  enc0_whh,  (size_t)G3 * HH);
  conv(enc1.w_ih,  enc1_wih,  (size_t)G3 * HH);  conv(enc1.w_hh,  enc1_whh,  (size_t)G3 * HH);
  conv(muW0, muW0b, (size_t)EHD * EHD); conv(muW1, muW1b, (size_t)LATD * EHD);
  conv(lvW0, lvW0b, (size_t)EHD * EHD); conv(lvW1, lvW1b, (size_t)LATD * EHD);
  conv(latW[0][0], latW0b[0], (size_t)LATD * LATD); conv(latW[0][1], latW1b[0], (size_t)HH * LATD);
  conv(latW[1][0], latW0b[1], (size_t)LATD * LATD); conv(latW[1][1], latW1b[1], (size_t)HH * LATD);
  conv(fcdW0, fcdW0b, (size_t)HH * HH); conv(fcdW1, fcdW1b, (size_t)NIN * HH);
  conv(x, xb, (size_t)BB * TT * NIN);

  auto gemm = [&](const unsigned short* A, int lda, const unsigned short* W, int K,
                  const float* bias, float* C, int ldc, unsigned short* Cb, int ldcb,
                  int M, int N, int act) {
    wmma_gemm_k<<<dim3((unsigned)(N / 64), (unsigned)(M / 64)), dim3(128), 0, stream>>>(
        A, lda, W, K, bias, C, ldc, Cb, ldcb, act);
  };
  // fused gi+gh gate GEMMs for one GRU layer (both N=G3, M=BB, act=0)
  auto gemm_gates = [&](const unsigned short* Ai, int ldai, const unsigned short* Wi, int Ki,
                        const float* bi,
                        const unsigned short* Ah, const unsigned short* Wh,
                        const float* bh) {
    wmma_gemm2_k<<<dim3(G3 / 64, BB / 64, 2), dim3(128), 0, stream>>>(
        Ai, ldai, Wi, Ki, bi, gi,
        Ah, HH,   Wh, HH, bh, gh, G3);
  };
  auto grupw = [&](float* h, unsigned short* hbp) {
    gru_pw_k<<<dim3(BB * HH / 256), dim3(256), 0, stream>>>(gi, gh, h, hbp);
  };
  auto zeroh = [&](int i) {
    zero_h_k<<<dim3(BB * HH / 256), dim3(256), 0, stream>>>(hf[i], hb[i], BB * HH);
  };

  // ---- forward encoder ----
  zeroh(0); zeroh(1);
  for (int t = 0; t < TT; ++t) {
    gemm_gates(xb + (size_t)t * NIN, TT * NIN, enc0_wih, NIN, enc0.b_ih,
               hb[0], enc0_whh, enc0.b_hh);
    grupw(hf[0], hb[0]);
    gemm_gates(hb[0], HH, enc1_wih, HH, enc1.b_ih,
               hb[1], enc1_whh, enc1.b_hh);
    grupw(hf[1], hb[1]);
  }
  // ---- backward encoder ----
  zeroh(2); zeroh(3);
  for (int t = TT - 1; t >= 0; --t) {
    gemm_gates(xb + (size_t)t * NIN, TT * NIN, benc0_wih, NIN, benc0.b_ih,
               hb[2], benc0_whh, benc0.b_hh);
    grupw(hf[2], hb[2]);
    gemm_gates(hb[2], HH, benc1_wih, HH, benc1.b_ih,
               hb[3], benc1_whh, benc1.b_hh);
    grupw(hf[3], hb[3]);
  }

  concat_hidden_k<<<dim3(BB * EHD / 256), dim3(256), 0, stream>>>(
      hb[0], hb[1], hb[2], hb[3], hiddenb);

  // ---- fc_mu / fc_logvar ----
  gemm(hiddenb, EHD, muW0b, EHD, muB0, fcmid, EHD, fcmidb, EHD, BB, EHD, 1);
  gemm(fcmidb, EHD, muW1b, EHD, muB1, out_mu, LATD, nullptr, 0, BB, LATD, 0);
  gemm(hiddenb, EHD, lvW0b, EHD, lvB0, fcmid, EHD, fcmidb, EHD, BB, EHD, 1);
  gemm(fcmidb, EHD, lvW1b, EHD, lvB1, out_lv, LATD, nullptr, 0, BB, LATD, 0);

  z_k<<<dim3(BB * LATD / 256), dim3(256), 0, stream>>>(out_mu, out_lv, eps, zf, zb);
  genre_k<<<dim3(1), dim3(256), 0, stream>>>(zf, out_g);

  // ---- fc_latent -> decoder initial hidden states ----
  for (int j = 0; j < 2; ++j) {
    gemm(zb, LATD, latW0b[j], LATD, latB[j][0], latmid, LATD, latmidb, LATD, BB, LATD, 1);
    gemm(latmidb, LATD, latW1b[j], LATD, latB[j][1], hf[4 + j], HH, hb[4 + j], HH, BB, HH, 0);
  }

  // ---- decoder (autoregressive) ----
  note0_k<<<dim3(BB * NIN / 256), dim3(256), 0, stream>>>(x, out, noteb);
  for (int t = 1; t < TT; ++t) {
    gemm_gates(noteb, NIN, dec0_wih, NIN, dec0.b_ih,
               hb[4], dec0_whh, dec0.b_hh);
    grupw(hf[4], hb[4]);
    gemm_gates(hb[4], HH, dec1_wih, HH, dec1.b_ih,
               hb[5], dec1_whh, dec1.b_hh);
    grupw(hf[5], hb[5]);
    gemm(hb[5], HH, fcdW0b, HH, fcdB0, fcdmid, HH, fcdmidb, HH, BB, HH, 1);
    gemm(fcdmidb, HH, fcdW1b, HH, fcdB1, out + (size_t)t * NIN, TT * NIN, noteb, NIN, BB, NIN, 0);
  }
}
